// SeparableConv_65309272703436
// MI455X (gfx1250) — compile-verified, matrix operand order checked by reading, not run on hardware
//
#include <hip/hip_runtime.h>

typedef __attribute__((ext_vector_type(16))) _Float16 v16h;
typedef __attribute__((ext_vector_type(8)))  float    v8f;
typedef __attribute__((ext_vector_type(4)))  float    v4f;

// One wave32 processes a tile of 16 rows with one V_WMMA_F32_16X16X32_F16.
// K index encodes (tap, channel): kappa = tap*8 + ch.
//   A[m,kappa] = w[tap] if ch==m else 0   (rows m=8..15 are zero)
//   B[kappa,n] = gathered[n, tap, ch]     (f16)
//   D[m,n]     = sum_tap w[tap] * gathered[n, tap, m]
// B VGPR layout (16-bit, 32x16): lanes 0-15 hold K=0..7 & 16..23 (taps 0,2),
// lanes 16-31 hold K=8..15 & 24..31 (taps 1,3) -> each lane gathers 2 rows
// of 8 contiguous floats (2x b128 loads). D: lane n (0..15), VGPR r = ch r.
__global__ __launch_bounds__(256) void sepconv_axis_wmma(
    const float* __restrict__ src, float* __restrict__ dst,
    const int* __restrict__ linear_idx, const int* __restrict__ table,
    const float* __restrict__ wk, int n_rows, int step)
{
  const int lane = threadIdx.x & 31;
  const int wave = threadIdx.x >> 5;
  const int tile = blockIdx.x * (blockDim.x >> 5) + wave;
  const int rowbase = tile * 16;
  if (rowbase >= n_rows) return;            // uniform per wave

  const int n  = lane & 15;                 // row within tile == A-row m
  const int hi = lane >> 4;                 // lane half selects tap pair

  // 4 tap weights (scalar-cached loads), converted once to f16
  const _Float16 w0 = (_Float16)wk[0], w1 = (_Float16)wk[1];
  const _Float16 w2 = (_Float16)wk[2], w3 = (_Float16)wk[3];
  const _Float16 wA = hi ? w1 : w0;         // tap index kA = hi
  const _Float16 wB = hi ? w3 : w2;         // tap index kB = hi + 2

  int ridx = rowbase + n;
  if (ridx > n_rows - 1) ridx = n_rows - 1; // tail clamp (keeps EXEC full)
  const int li = linear_idx[ridx];

  // Gather the two tap rows this lane owns; branchless validity select.
  v16h b;
  #pragma unroll
  for (int t = 0; t < 2; ++t) {
    const int k = (t == 0) ? hi : (hi + 2);          // kernel tap index 0..3
    const long nbr = (long)li + (long)step * (k - 2); // tap offsets -2..+1
    const bool valid = (nbr >= 0) && (nbr < (long)n_rows);
    long cl = nbr;
    if (cl < 0) cl = 0;
    if (cl > (long)n_rows - 1) cl = (long)n_rows - 1;
    const int row = table[cl];
    const float* g = src + (size_t)row * 8;
    const v4f g0 = *(const v4f*)(g);                 // global_load_b128
    const v4f g1 = *(const v4f*)(g + 4);             // global_load_b128
    #pragma unroll
    for (int c4 = 0; c4 < 4; ++c4) {
      b[t * 8 + c4]     = valid ? (_Float16)g0[c4] : (_Float16)0.0f;
      b[t * 8 + 4 + c4] = valid ? (_Float16)g1[c4] : (_Float16)0.0f;
    }
  }

  // Sparse weight matrix A in the 16-bit 16x32 A layout.
  v16h a;
  #pragma unroll
  for (int j = 0; j < 8; ++j) {
    a[j]     = (j == n) ? wA : (_Float16)0.0f;
    a[j + 8] = (j == n) ? wB : (_Float16)0.0f;
  }

  v8f c = {};
  c = __builtin_amdgcn_wmma_f32_16x16x32_f16(
      /*neg_a=*/false, a, /*neg_b=*/false, b,
      /*c_mod=*/(short)0, c, /*reuse_a=*/false, /*reuse_b=*/false);

  // Lanes 0-15 hold channels 0..7 (c[0..7]) of row n -> one 32B store.
  if (hi == 0 && (rowbase + n) < n_rows) {
    float* o = dst + (size_t)(rowbase + n) * 8;
    v4f o0 = { c[0], c[1], c[2], c[3] };
    v4f o1 = { c[4], c[5], c[6], c[7] };
    *(v4f*)(o)     = o0;
    *(v4f*)(o + 4) = o1;
  }
}

extern "C" void kernel_launch(void* const* d_in, const int* in_sizes, int n_in,
                              void* d_out, int out_size, void* d_ws, size_t ws_size,
                              hipStream_t stream) {
  // setup_inputs() dict order:
  // 0: signal f32[N*8]  1: linear_idx i32[N]  2: table i32[N]
  // 3: grid_size (scalar)  4..6: kernel0..2 f32[4]
  const float* signal     = (const float*)d_in[0];
  const int*   linear_idx = (const int*)  d_in[1];
  const int*   table      = (const int*)  d_in[2];
  const float* k0         = (const float*)d_in[4];
  const float* k1         = (const float*)d_in[5];
  const float* k2         = (const float*)d_in[6];
  float* out = (float*)d_out;
  float* ws  = (float*)d_ws;    // 64 MB intermediate (N*8 f32)

  const int N = in_sizes[1];
  // G = cbrt(N) (full occupancy grid); integer search, tiny.
  int G = 1;
  while ((long)G * G * G < (long)N) ++G;

  // step_i = offset_i * dilation_i with offsets G^2, G, 1 and DIL = (1,2,4)
  const int step0 = G * G * 1;
  const int step1 = G * 2;
  const int step2 = 1 * 4;

  const int tiles  = (N + 15) / 16;
  const int blocks = (tiles + 7) / 8;   // 8 waves (tiles) per 256-thread block

  // Ping-pong so every pass reads a fully-written buffer:
  // signal -> out -> ws -> out. Working set (~80 MB) stays L2-resident.
  sepconv_axis_wmma<<<blocks, 256, 0, stream>>>(signal, out, linear_idx, table, k0, N, step0);
  sepconv_axis_wmma<<<blocks, 256, 0, stream>>>(out,    ws,  linear_idx, table, k1, N, step1);
  sepconv_axis_wmma<<<blocks, 256, 0, stream>>>(ws,     out, linear_idx, table, k2, N, step2);
}
